// APPNPNet_18038862643740
// MI455X (gfx1250) — compile-verified, matrix operand order checked by reading, not run on hardware
//
#include <hip/hip_runtime.h>
#include <stdint.h>

// ---------------------------------------------------------------------------
// APPNP on MI455X (gfx1250):
//   h  = relu(x@W1+b1); h = relu(h@W2+b2); h = h@W3+b3    (WMMA bf16 GEMMs,
//        B panels staged in LDS via async global->LDS copies)
//   CSR-transpose build on device, then 10x atomic-free gather iterations:
//   z_{k+1}[d] = 0.9 * sum_e w_e * z_k[src_e] + 0.1 * h[d]
// ---------------------------------------------------------------------------

typedef __attribute__((ext_vector_type(16))) __bf16 v16bf;
typedef __attribute__((ext_vector_type(8)))  float  v8f;
typedef __attribute__((ext_vector_type(4)))  int    v4i;

static __device__ __forceinline__ uint16_t f32_to_bf16(float f) {
    uint32_t u = __float_as_uint(f);
    uint32_t r = (u + 0x7FFFu + ((u >> 16) & 1u)) >> 16;   // round-to-nearest-even
    return (uint16_t)r;
}
static __device__ __forceinline__ uint32_t pack_bf16x2(float lo, float hi) {
    return (uint32_t)f32_to_bf16(lo) | ((uint32_t)f32_to_bf16(hi) << 16);
}

union Frag16 {           // one WMMA 16-bit operand: 16 bf16 = 8 VGPRs per lane
    uint32_t u[8];
    uint4    q[2];
    v16bf    v;
};

// ---- gfx1250 async global->LDS staging (ASYNCcnt path), with fallback -----
static __device__ __forceinline__ void stage_b128(const uint4* g, uint4* l) {
#if __has_builtin(__builtin_amdgcn_global_load_async_to_lds_b128)
    // prototype: (v4i addrspace(1)*, v4i addrspace(3)*, int offset, int cpol)
    __builtin_amdgcn_global_load_async_to_lds_b128(
        (__attribute__((address_space(1))) v4i*)g,
        (__attribute__((address_space(3))) v4i*)l, 0, 0);
#else
    *l = *g;                    // sync fallback: global_load + ds_store
#endif
}
template <int N>
static __device__ __forceinline__ void wait_async() {
#if __has_builtin(__builtin_amdgcn_s_wait_asynccnt)
    __builtin_amdgcn_s_wait_asynccnt((short)N);
#else
    asm volatile("s_wait_asynccnt %0" ::"n"(N) : "memory");
#endif
}

// ---------------------------------------------------------------------------
// Pre-pack weights [K,N] f32 row-major -> bf16 in WMMA B-fragment order:
//   out[(((nt*KS + ks)*32 + lane)*16 + j)] = bf16(W[k, n])
//   n = nt*16 + (lane&15);  k = ks*32 + j + (j<8?0:8) + (lane<16?0:8)
// ---------------------------------------------------------------------------
__global__ void pack_w_kernel(const float* __restrict__ W, uint16_t* __restrict__ out,
                              int K, int N) {
    int t = blockIdx.x * blockDim.x + threadIdx.x;
    if (t >= K * N) return;
    int j    = t & 15;
    int lane = (t >> 4) & 31;
    int rest = t >> 9;
    int KS   = K >> 5;
    int ks   = rest % KS;
    int nt   = rest / KS;
    int n = nt * 16 + (lane & 15);
    int k = ks * 32 + j + ((j < 8) ? 0 : 8) + ((lane < 16) ? 0 : 8);
    out[t] = f32_to_bf16(W[(size_t)k * N + n]);
}

// ---------------------------------------------------------------------------
// WMMA GEMM: C[M, NT*16] = act(A[M, KSTEPS*32] @ B + bias)
//  - 4 waves/block, one 16-row strip per wave; A streamed exactly once
//  - B panel (NT KB per K-step) async-staged into LDS, double-buffered,
//    shared by all 4 waves; WMMA operands come from ds_load_b128
// ---------------------------------------------------------------------------
template <int NT, int KSTEPS, bool A_F32, bool RELU, bool OUT_BF16>
__global__ __launch_bounds__(128)
void gemm_wmma(const void* __restrict__ Aptr, const uint16_t* __restrict__ Bp,
               const float* __restrict__ bias, void* __restrict__ Cptr, int M) {
    constexpr int K = KSTEPS * 32;
    constexpr int N = NT * 16;
    constexpr int CHUNKS = NT * 64;       // 16B chunks per K-step panel
    constexpr int PT = CHUNKS / 128;      // async issues per thread per stage

    __shared__ uint4 panel[2][CHUNKS];

    const int tid  = threadIdx.x;
    const int lane = tid & 31;
    const int wave = tid >> 5;
    const int R = (blockIdx.x * 4 + wave) * 16;
    const bool active = (R < M);          // wave-uniform; inactive waves still stage

    const int aRow = R + (lane & 15);
    const int koff = (lane < 16) ? 0 : 8; // ISA 16-bit A-fragment K offset

    const float*    af32 = (const float*)Aptr    + (size_t)aRow * K;
    const uint16_t* abf  = (const uint16_t*)Aptr + (size_t)aRow * K;
    const uint4*    Bp4  = (const uint4*)Bp;

    v8f acc[NT] = {};

    // prologue: stage panel for ks=0
#pragma unroll
    for (int q = tid; q < CHUNKS; q += 128) {
        int nt = q >> 6, ch = q & 63;
        stage_b128(Bp4 + ((size_t)(nt * KSTEPS + 0) * 64 + ch), &panel[0][q]);
    }

#pragma unroll 1
    for (int ks = 0; ks < KSTEPS; ++ks) {
        const int k0 = ks * 32;

        // A fragment for this K-step (overlaps with barrier / async copies)
        Frag16 a;
        if (active) {
            if (A_F32) {
                const float4 f0 = *(const float4*)(af32 + k0 + koff);
                const float4 f1 = *(const float4*)(af32 + k0 + koff + 4);
                const float4 f2 = *(const float4*)(af32 + k0 + 16 + koff);
                const float4 f3 = *(const float4*)(af32 + k0 + 16 + koff + 4);
                a.u[0] = pack_bf16x2(f0.x, f0.y);
                a.u[1] = pack_bf16x2(f0.z, f0.w);
                a.u[2] = pack_bf16x2(f1.x, f1.y);
                a.u[3] = pack_bf16x2(f1.z, f1.w);
                a.u[4] = pack_bf16x2(f2.x, f2.y);
                a.u[5] = pack_bf16x2(f2.z, f2.w);
                a.u[6] = pack_bf16x2(f3.x, f3.y);
                a.u[7] = pack_bf16x2(f3.z, f3.w);
                if (ks + 1 < KSTEPS) __builtin_prefetch(af32 + k0 + 32, 0, 0);
            } else {
                a.q[0] = *(const uint4*)(abf + k0 + koff);
                a.q[1] = *(const uint4*)(abf + k0 + 16 + koff);
                if (ks + 1 < KSTEPS) __builtin_prefetch(abf + k0 + 32, 0, 0);
            }
        }

        __syncthreads();   // everyone done reading panel[(ks+1)&1] from step ks-1
        if (ks + 1 < KSTEPS) {
#pragma unroll
            for (int q = tid; q < CHUNKS; q += 128) {
                int nt = q >> 6, ch = q & 63;
                stage_b128(Bp4 + ((size_t)(nt * KSTEPS + (ks + 1)) * 64 + ch),
                           &panel[(ks + 1) & 1][q]);
            }
            wait_async<PT>();   // my chunks of panel[ks&1] complete; prefetch in flight
        } else {
            wait_async<0>();
        }
        __syncthreads();   // all waves' chunks of panel[ks&1] complete

        if (active) {
            const uint4* pan = &panel[ks & 1][lane * 2];
#pragma unroll
            for (int nt = 0; nt < NT; ++nt) {
                Frag16 b;
                b.q[0] = pan[nt * 64];        // ds_load_b128
                b.q[1] = pan[nt * 64 + 1];    // ds_load_b128
                acc[nt] = __builtin_amdgcn_wmma_f32_16x16x32_bf16(
                    false, a.v, false, b.v, (short)0, acc[nt], false, false);
            }
        }
    }

    if (!active) return;

    // Epilogue: C/D layout (ISA §7.12.2): VGPR r -> row r / r+8, lane%16 -> col
    const int colLane = lane & 15;
    const int rBase   = R + ((lane < 16) ? 0 : 8);
#pragma unroll
    for (int nt = 0; nt < NT; ++nt) {
        const int col = nt * 16 + colLane;
        const float bv = bias[col];
#pragma unroll
        for (int r = 0; r < 8; ++r) {
            float v = acc[nt][r] + bv;
            if (RELU) v = fmaxf(v, 0.0f);
            if (OUT_BF16)
                ((uint16_t*)Cptr)[(size_t)(rBase + r) * N + col] = f32_to_bf16(v);
            else
                ((float*)Cptr)[(size_t)(rBase + r) * N + col] = v;
        }
    }
}

// ---------------------------------------------------------------------------
// CSR build (transpose graph: group edges by destination)
// ---------------------------------------------------------------------------
__global__ void zero_int_kernel(int* __restrict__ p, int n) {
    int i = blockIdx.x * blockDim.x + threadIdx.x;
    if (i < n) p[i] = 0;
}
__global__ void copy_int_kernel(const int* __restrict__ a, int* __restrict__ b, int n) {
    int i = blockIdx.x * blockDim.x + threadIdx.x;
    if (i < n) b[i] = a[i];
}
__global__ void count_deg_kernel(const int* __restrict__ dst, int* __restrict__ cnt, int nE) {
    int e = blockIdx.x * blockDim.x + threadIdx.x;
    if (e < nE) atomicAdd(&cnt[dst[e]], 1);
}
// single-block exclusive scan (n ~ 100K: a few microseconds, fully deterministic)
__global__ __launch_bounds__(1024)
void exclusive_scan_kernel(const int* __restrict__ in, int* __restrict__ out, int n) {
    __shared__ int buf[1024];
    __shared__ int carry;
    const int tid = threadIdx.x;
    if (tid == 0) carry = 0;
    __syncthreads();
    for (int base = 0; base < n; base += 1024) {
        const int i = base + tid;
        const int v = (i < n) ? in[i] : 0;
        buf[tid] = v;
        __syncthreads();
#pragma unroll
        for (int ofs = 1; ofs < 1024; ofs <<= 1) {
            int t = (tid >= ofs) ? buf[tid - ofs] : 0;
            __syncthreads();
            buf[tid] += t;
            __syncthreads();
        }
        const int c = carry;
        if (i < n) out[i] = c + buf[tid] - v;   // exclusive
        __syncthreads();
        if (tid == 1023) carry = c + buf[1023];
        __syncthreads();
    }
    if (tid == 0) out[n] = carry;
}
__global__ void csr_fill_kernel(const int* __restrict__ src, const int* __restrict__ dst,
                                const float* __restrict__ w, int* __restrict__ cursor,
                                int* __restrict__ csr_src, float* __restrict__ csr_w, int nE) {
    int e = blockIdx.x * blockDim.x + threadIdx.x;
    if (e >= nE) return;
    int pos = atomicAdd(&cursor[dst[e]], 1);
    csr_src[pos] = src[e];
    csr_w[pos]   = w[e];
}

// ---------------------------------------------------------------------------
// One APPNP iteration, atomic-free gather form (z rows are L2-resident):
//   out[d] = 0.9 * sum_{e in CSR[d]} w_e * z[src_e] + 0.1 * h[d]
// 16 lanes per node, one float4 of channels per lane -> 256B contiguous gathers
// ---------------------------------------------------------------------------
__global__ __launch_bounds__(256)
void appnp_gather_kernel(const float* __restrict__ z, const float* __restrict__ h,
                         const int* __restrict__ rowptr, const int* __restrict__ csr_src,
                         const float* __restrict__ csr_w, float* __restrict__ out,
                         int nNodes) {
    int t = blockIdx.x * blockDim.x + threadIdx.x;
    int node = t >> 4;
    if (node >= nNodes) return;
    const int c = (t & 15) << 2;
    const int beg = rowptr[node];
    const int end = rowptr[node + 1];
    float4 acc = make_float4(0.f, 0.f, 0.f, 0.f);
    int e = beg;
    for (; e + 1 < end; e += 2) {                 // 2-edge ILP
        const int   s0 = csr_src[e],     s1 = csr_src[e + 1];
        const float w0 = csr_w[e],       w1 = csr_w[e + 1];
        const float4 v0 = *(const float4*)(z + (size_t)s0 * 64 + c);
        const float4 v1 = *(const float4*)(z + (size_t)s1 * 64 + c);
        acc.x += w0 * v0.x + w1 * v1.x;
        acc.y += w0 * v0.y + w1 * v1.y;
        acc.z += w0 * v0.z + w1 * v1.z;
        acc.w += w0 * v0.w + w1 * v1.w;
    }
    if (e < end) {
        const int   s0 = csr_src[e];
        const float w0 = csr_w[e];
        const float4 v0 = *(const float4*)(z + (size_t)s0 * 64 + c);
        acc.x += w0 * v0.x; acc.y += w0 * v0.y; acc.z += w0 * v0.z; acc.w += w0 * v0.w;
    }
    const float4 hv = *(const float4*)(h + (size_t)node * 64 + c);
    float4 o;
    o.x = 0.9f * acc.x + 0.1f * hv.x;
    o.y = 0.9f * acc.y + 0.1f * hv.y;
    o.z = 0.9f * acc.z + 0.1f * hv.z;
    o.w = 0.9f * acc.w + 0.1f * hv.w;
    *(float4*)(out + (size_t)node * 64 + c) = o;
}

// ---------------------------------------------------------------------------
extern "C" void kernel_launch(void* const* d_in, const int* in_sizes, int n_in,
                              void* d_out, int out_size, void* d_ws, size_t ws_size,
                              hipStream_t stream) {
    const float* x  = (const float*)d_in[0];
    const float* W1 = (const float*)d_in[1];
    const float* b1 = (const float*)d_in[2];
    const float* W2 = (const float*)d_in[3];
    const float* b2 = (const float*)d_in[4];
    const float* W3 = (const float*)d_in[5];
    const float* b3 = (const float*)d_in[6];
    const float* ew = (const float*)d_in[7];
    const int* esrc = (const int*)d_in[8];
    const int* edst = (const int*)d_in[9];

    constexpr int NN = 100000, E = 3200000;
    constexpr int IN_CH = 512, HID = 256, OUT = 64;

    char* ws = (char*)d_ws;
    size_t off = 0;
    auto alloc = [&](size_t bytes) {
        void* p = ws + off;
        off += (bytes + 255) & ~(size_t)255;
        return p;
    };
    uint16_t* h1   = (uint16_t*)alloc((size_t)NN * HID * 2);   // 51.2 MB (reused for CSR)
    uint16_t* h2   = (uint16_t*)alloc((size_t)NN * HID * 2);   // 51.2 MB
    float*    hf   = (float*)   alloc((size_t)NN * OUT * 4);   // 25.6 MB
    float*    bufA = (float*)   alloc((size_t)NN * OUT * 4);   // 25.6 MB
    float*    bufB = (float*)   alloc((size_t)NN * OUT * 4);   // 25.6 MB
    uint16_t* W1p  = (uint16_t*)alloc((size_t)IN_CH * HID * 2);
    uint16_t* W2p  = (uint16_t*)alloc((size_t)HID * HID * 2);
    uint16_t* W3p  = (uint16_t*)alloc((size_t)HID * OUT * 2);

    // CSR arrays overlaid on h1 (free after GEMM2; stream order serializes reuse)
    int*   rowptr  = (int*)(void*)h1;          // NN+1
    int*   cursor  = rowptr + (NN + 1);        // NN
    int*   csr_src = cursor + NN;              // E
    float* csr_w   = (float*)(csr_src + E);    // E   (total 26.4 MB <= 51.2 MB)

    // 1) pack weights into WMMA B-fragment order (bf16)
    pack_w_kernel<<<(IN_CH * HID + 255) / 256, 256, 0, stream>>>(W1, W1p, IN_CH, HID);
    pack_w_kernel<<<(HID * HID + 255) / 256, 256, 0, stream>>>(W2, W2p, HID, HID);
    pack_w_kernel<<<(HID * OUT + 255) / 256, 256, 0, stream>>>(W3, W3p, HID, OUT);

    // 2) MLP encoder via WMMA + async-LDS-staged B panels
    const int gBlocks = (NN / 16 + 3) / 4;     // 1563
    gemm_wmma<16, 16, true,  true,  true ><<<gBlocks, 128, 0, stream>>>(x,  W1p, b1, h1, NN);
    gemm_wmma<16, 8,  false, true,  true ><<<gBlocks, 128, 0, stream>>>(h1, W2p, b2, h2, NN);
    gemm_wmma<4,  8,  false, false, false><<<gBlocks, 128, 0, stream>>>(h2, W3p, b3, hf, NN);

    // 3) CSR-transpose build (after GEMM2 so h1's memory is free)
    zero_int_kernel<<<(NN + 255) / 256, 256, 0, stream>>>(cursor, NN);
    count_deg_kernel<<<(E + 255) / 256, 256, 0, stream>>>(edst, cursor, E);
    exclusive_scan_kernel<<<1, 1024, 0, stream>>>(cursor, rowptr, NN);
    copy_int_kernel<<<(NN + 255) / 256, 256, 0, stream>>>(rowptr, cursor, NN);
    csr_fill_kernel<<<(E + 255) / 256, 256, 0, stream>>>(esrc, edst, ew, cursor,
                                                         csr_src, csr_w, E);

    // 4) 10 propagation iterations, gather form, ping-pong; last writes d_out
    const float* cur = hf;
    const int gatherBlocks = (NN * 16 + 255) / 256;
    for (int k = 0; k < 10; ++k) {
        float* next = (k == 9) ? (float*)d_out : ((k & 1) ? bufB : bufA);
        appnp_gather_kernel<<<gatherBlocks, 256, 0, stream>>>(cur, hf, rowptr,
                                                              csr_src, csr_w, next, NN);
        cur = next;
    }
}
